// PointTransformerBlock_40819369181270
// MI455X (gfx1250) — matrix operand order repaired and checked by reading, not compiled, not run on hardware
//
#include <hip/hip_runtime.h>
#include <hip/hip_bf16.h>

#define BB 2
#define NN 8192
#define CC 64
#define KK 16
#define EPSBN 1e-5f

typedef __attribute__((ext_vector_type(16))) _Float16 hv16;
typedef __attribute__((ext_vector_type(8)))  _Float16 hv8;
typedef __attribute__((ext_vector_type(8)))  float    fv8;

#define WMMA_F16(A, Bf, Cc) \
  __builtin_amdgcn_wmma_f32_16x16x32_f16(false, (A), false, (Bf), (short)0, (Cc), false, false)

// ---------------------------------------------------------------------------
// Fragment helpers (layouts per CDNA5 ISA 7.12.2, wave32)
// ---------------------------------------------------------------------------

// Pre-shuffle a 64x64 row-major f32 weight matrix W[k][n] into LDS "fragment
// order": frag = kt*4+nt, each lane's 16 halves contiguous (32B -> 1 vector ld).
__device__ __forceinline__ void fill_wfrag(_Float16* dst, const float* __restrict__ W) {
  for (int idx = threadIdx.x; idx < 4096; idx += blockDim.x) {
    int h    = idx & 15;
    int lane = (idx >> 4) & 31;
    int frag = idx >> 9;                 // 0..7
    int kt = frag >> 2, nt = frag & 3;
    int k = kt * 32 + (lane >> 4) * 8 + (h & 7) + (h >> 3) * 16;   // B 32x16 layout: lanes<16 K=0..7,16..23
    int n = nt * 16 + (lane & 15);
    dst[idx] = (_Float16)W[k * CC + n];
  }
}

__device__ __forceinline__ hv16 ld_b_frag(const _Float16* wf, int kt, int nt) {
  const int lane = threadIdx.x & 31;
  return *(const hv16*)(wf + (((kt * 4 + nt) * 32) + lane) * 16);
}

// A fragment (16x32 f16, M x K) from a row-major 16x64 f16 LDS tile.
__device__ __forceinline__ hv16 ld_a_lds(const _Float16* tile, int k0) {
  const int lane = threadIdx.x & 31;
  const int m = lane & 15, sel = lane >> 4;
  const _Float16* pb = tile + m * CC + k0 + sel * 8;
  hv8 lo = *(const hv8*)(pb);        // K = k0 + sel*8 + 0..7
  hv8 hi = *(const hv8*)(pb + 16);   // K = k0 + 16 + sel*8 + 0..7
  hv16 r;
#pragma unroll
  for (int t = 0; t < 8; ++t) { r[t] = lo[t]; r[8 + t] = hi[t]; }
  return r;
}

// A fragment straight from a row-major f32 global matrix (convert to f16).
__device__ __forceinline__ hv16 ld_a_gmem(const float* __restrict__ src, int rt0, int k0) {
  const int lane = threadIdx.x & 31;
  const int m = lane & 15, sel = lane >> 4;
  const float* rp = src + (size_t)(rt0 + m) * CC + k0 + sel * 8;
  hv16 r;
#pragma unroll
  for (int t = 0; t < 8; ++t) {
    r[t]     = (_Float16)rp[t];
    r[8 + t] = (_Float16)rp[16 + t];
  }
  return r;
}

// ---------------------------------------------------------------------------
// Kernel 1: brute-force KNN (drop self) + passthrough p
// ---------------------------------------------------------------------------
__global__ void __launch_bounds__(256)
knn_kernel(const float* __restrict__ p, int* __restrict__ ind, float* __restrict__ pout) {
  __shared__ float tile[256 * 3];
  const int tid  = threadIdx.x;
  const int bper = NN / 256;
  const int b    = blockIdx.x / bper;
  const int i    = (blockIdx.x % bper) * 256 + tid;
  const float* pb = p + (size_t)b * NN * 3;
  const float qx = pb[i * 3], qy = pb[i * 3 + 1], qz = pb[i * 3 + 2];

  float dlist[17]; int ilist[17];
#pragma unroll
  for (int s = 0; s < 17; ++s) { dlist[s] = 3.4e38f; ilist[s] = -1; }

  for (int t = 0; t < NN / 256; ++t) {
    __syncthreads();
    for (int idx = tid; idx < 768; idx += 256) tile[idx] = pb[t * 768 + idx];
    __syncthreads();
    if (t + 1 < NN / 256) __builtin_prefetch(pb + (t + 1) * 768 + tid, 0, 1);
    for (int j = 0; j < 256; ++j) {
      float dx = qx - tile[j * 3 + 0];
      float dy = qy - tile[j * 3 + 1];
      float dz = qz - tile[j * 3 + 2];
      float d  = dx * dx + dy * dy + dz * dz;
      if (d < dlist[16]) {               // sorted-insert (strict < keeps tie order)
        float dc = d; int ic = t * 256 + j;
#pragma unroll
        for (int s = 0; s < 17; ++s) {
          if (dc < dlist[s]) {
            float dt = dlist[s]; int it = ilist[s];
            dlist[s] = dc; ilist[s] = ic; dc = dt; ic = it;
          }
        }
      }
    }
  }
  int* dst = ind + ((size_t)b * NN + i) * KK;
#pragma unroll
  for (int s = 1; s < 17; ++s) dst[s - 1] = ilist[s];   // drop self (rank 0)
  float* po = pout + ((size_t)b * NN + i) * 3;
  po[0] = qx; po[1] = qy; po[2] = qz;
}

// ---------------------------------------------------------------------------
// Kernel 2: h = relu(bn1(x@w1));  q/k/v = h@W + b   (one wave = 16 rows)
// ---------------------------------------------------------------------------
__global__ void __launch_bounds__(256)
qkv_kernel(const float* __restrict__ x,
           const float* __restrict__ w1, const float* __restrict__ bn1_g, const float* __restrict__ bn1_b,
           const float* __restrict__ wq, const float* __restrict__ bq,
           const float* __restrict__ wk, const float* __restrict__ bk,
           const float* __restrict__ wv, const float* __restrict__ bv,
           float* __restrict__ qb, float* __restrict__ kb, float* __restrict__ vb) {
  __shared__ alignas(32) _Float16 wf[4][4096];     // 32 KB
  __shared__ alignas(32) _Float16 ht[8][16 * CC];  // 16 KB
  fill_wfrag(wf[0], w1); fill_wfrag(wf[1], wq);
  fill_wfrag(wf[2], wk); fill_wfrag(wf[3], wv);
  __syncthreads();

  const int wave = threadIdx.x >> 5, lane = threadIdx.x & 31;
  const int m = lane & 15, sel = lane >> 4;
  const int rt0 = (blockIdx.x * 8 + wave) * 16;
  const float rs = rsqrtf(1.0f + EPSBN);

  hv16 xa0 = ld_a_gmem(x, rt0, 0);
  hv16 xa1 = ld_a_gmem(x, rt0, 32);

#pragma unroll
  for (int nt = 0; nt < 4; ++nt) {
    fv8 acc = {};
    acc = WMMA_F16(xa0, ld_b_frag(wf[0], 0, nt), acc);
    acc = WMMA_F16(xa1, ld_b_frag(wf[0], 1, nt), acc);
    const int c = nt * 16 + m;
    const float s1 = bn1_g[c] * rs, o1 = bn1_b[c];
#pragma unroll
    for (int r = 0; r < 8; ++r) {
      float v = fmaf(acc[r], s1, o1);
      ht[wave][(r + sel * 8) * CC + c] = (_Float16)fmaxf(v, 0.0f);
    }
  }
  __syncthreads();

  hv16 ha0 = ld_a_lds(ht[wave], 0);
  hv16 ha1 = ld_a_lds(ht[wave], 32);

  const _Float16* wm3[3] = { wf[1], wf[2], wf[3] };
  const float*    bv3[3] = { bq, bk, bv };
  float*          ob3[3] = { qb, kb, vb };
#pragma unroll
  for (int mm = 0; mm < 3; ++mm) {
#pragma unroll
    for (int nt = 0; nt < 4; ++nt) {
      fv8 acc = {};
      acc = WMMA_F16(ha0, ld_b_frag(wm3[mm], 0, nt), acc);
      acc = WMMA_F16(ha1, ld_b_frag(wm3[mm], 1, nt), acc);
      const int c = nt * 16 + m;
      const float bias = bv3[mm][c];
      float* dst = ob3[mm] + (size_t)rt0 * CC;
#pragma unroll
      for (int r = 0; r < 8; ++r) dst[(r + sel * 8) * CC + c] = acc[r] + bias;
    }
  }
}

// ---------------------------------------------------------------------------
// Kernel 3: fused pos-MLP + vector attention (one wave = one query, 4 waves/blk)
// ---------------------------------------------------------------------------
__global__ void __launch_bounds__(128)
attn_kernel(const float* __restrict__ p, const int* __restrict__ ind,
            const float* __restrict__ qb, const float* __restrict__ kb, const float* __restrict__ vb,
            const float* __restrict__ lp1_w, const float* __restrict__ lp1_b,
            const float* __restrict__ bnp_g, const float* __restrict__ bnp_b,
            const float* __restrict__ lp2_w, const float* __restrict__ lp2_b,
            const float* __restrict__ bnw1_g, const float* __restrict__ bnw1_b,
            const float* __restrict__ lw1_w,  const float* __restrict__ lw1_b,
            const float* __restrict__ bnw2_g, const float* __restrict__ bnw2_b,
            const float* __restrict__ lw2_w,  const float* __restrict__ lw2_b,
            float* __restrict__ hout) {
  __shared__ alignas(32) _Float16 lwf[2][4096];     // lw1, lw2 : 16 KB
  __shared__ alignas(32) _Float16 wm[4][16 * CC];   //  8 KB
  __shared__ float vpr[4][16 * CC];                 // 16 KB
  __shared__ float pnb[4][52];
  __shared__ int   nid[4][16];
  fill_wfrag(lwf[0], lw1_w); fill_wfrag(lwf[1], lw2_w);

  const int wave = threadIdx.x >> 5, lane = threadIdx.x & 31;
  const int qi = blockIdx.x * 4 + wave;          // flattened (b,n)
  const int b = qi / NN, i = qi - b * NN;

  if (lane < KK) {
    int j = ind[(size_t)qi * KK + lane];
    nid[wave][lane] = j;
    const float* pj = p + ((size_t)b * NN + j) * 3;
    pnb[wave][lane * 3 + 0] = pj[0];
    pnb[wave][lane * 3 + 1] = pj[1];
    pnb[wave][lane * 3 + 2] = pj[2];
  } else if (lane < KK + 3) {
    pnb[wave][48 + (lane - KK)] = p[((size_t)b * NN + i) * 3 + (lane - KK)];
  }
  __syncthreads();

  const float rs = rsqrtf(1.0f + EPSBN);
  const int c0 = lane * 2;                        // each lane owns 2 channels
  const float2 qv  = *(const float2*)(qb + (size_t)qi * CC + c0);
  const float2 l20 = *(const float2*)(lp2_w + 0 * CC + c0);
  const float2 l21 = *(const float2*)(lp2_w + 1 * CC + c0);
  const float2 l22 = *(const float2*)(lp2_w + 2 * CC + c0);
  const float2 l2b = *(const float2*)(lp2_b + c0);
  const float2 g1  = *(const float2*)(bnw1_g + c0);
  const float2 b1  = *(const float2*)(bnw1_b + c0);
  const float sA = g1.x * rs, sB = g1.y * rs;

  float w9[9], pb3[3], spg[3], spb[3];
#pragma unroll
  for (int t = 0; t < 9; ++t) w9[t] = lp1_w[t];
#pragma unroll
  for (int t = 0; t < 3; ++t) { pb3[t] = lp1_b[t]; spg[t] = bnp_g[t] * rs; spb[t] = bnp_b[t]; }
  const float pix = pnb[wave][48], piy = pnb[wave][49], piz = pnb[wave][50];

#pragma unroll
  for (int j16 = 0; j16 < KK; ++j16) {
    float dx = pnb[wave][j16 * 3 + 0] - pix;
    float dy = pnb[wave][j16 * 3 + 1] - piy;
    float dz = pnb[wave][j16 * 3 + 2] - piz;
    float e0 = dx * w9[0] + dy * w9[3] + dz * w9[6] + pb3[0];
    float e1 = dx * w9[1] + dy * w9[4] + dz * w9[7] + pb3[1];
    float e2 = dx * w9[2] + dy * w9[5] + dz * w9[8] + pb3[2];
    e0 = fmaxf(fmaf(e0, spg[0], spb[0]), 0.0f);
    e1 = fmaxf(fmaf(e1, spg[1], spb[1]), 0.0f);
    e2 = fmaxf(fmaf(e2, spg[2], spb[2]), 0.0f);
    float pr0 = e0 * l20.x + e1 * l21.x + e2 * l22.x + l2b.x;
    float pr1 = e0 * l20.y + e1 * l21.y + e2 * l22.y + l2b.y;
    const int j = nid[wave][j16];
    const float2 kv2 = *(const float2*)(kb + ((size_t)b * NN + j) * CC + c0);
    const float2 vv2 = *(const float2*)(vb + ((size_t)b * NN + j) * CC + c0);
    float w0 = fmaxf(fmaf(qv.x - kv2.x + pr0, sA, b1.x), 0.0f);
    float w1 = fmaxf(fmaf(qv.y - kv2.y + pr1, sB, b1.y), 0.0f);
    wm[wave][j16 * CC + c0]     = (_Float16)w0;
    wm[wave][j16 * CC + c0 + 1] = (_Float16)w1;
    vpr[wave][j16 * CC + c0]     = vv2.x + pr0;
    vpr[wave][j16 * CC + c0 + 1] = vv2.y + pr1;
  }
  __syncthreads();

  const int m = lane & 15, sel = lane >> 4;
  {  // GEMM 1: wm @ lw1 + b -> bnw2 relu -> wm (f16)
    hv16 a0 = ld_a_lds(wm[wave], 0);
    hv16 a1 = ld_a_lds(wm[wave], 32);
#pragma unroll
    for (int nt = 0; nt < 4; ++nt) {
      fv8 acc = {};
      acc = WMMA_F16(a0, ld_b_frag(lwf[0], 0, nt), acc);
      acc = WMMA_F16(a1, ld_b_frag(lwf[0], 1, nt), acc);
      const int c = nt * 16 + m;
      const float bias = lw1_b[c];
      const float s2 = bnw2_g[c] * rs, o2 = bnw2_b[c];
#pragma unroll
      for (int r = 0; r < 8; ++r) {
        float v = fmaxf(fmaf(acc[r] + bias, s2, o2), 0.0f);
        wm[wave][(r + sel * 8) * CC + c] = (_Float16)v;
      }
    }
  }
  __syncthreads();
  {  // GEMM 2 + softmax over neighbors + weighted sum
    hv16 a0 = ld_a_lds(wm[wave], 0);
    hv16 a1 = ld_a_lds(wm[wave], 32);
#pragma unroll
    for (int nt = 0; nt < 4; ++nt) {
      fv8 acc = {};
      acc = WMMA_F16(a0, ld_b_frag(lwf[1], 0, nt), acc);
      acc = WMMA_F16(a1, ld_b_frag(lwf[1], 1, nt), acc);
      const int c = nt * 16 + m;
      const float bias = lw2_b[c];
      float mx = -3.4e38f;
#pragma unroll
      for (int r = 0; r < 8; ++r) { acc[r] += bias; mx = fmaxf(mx, acc[r]); }
      mx = fmaxf(mx, __shfl_xor(mx, 16, 32));     // rows 0-7 <-> 8-15
      float ex[8], sum = 0.0f;
#pragma unroll
      for (int r = 0; r < 8; ++r) { ex[r] = __expf(acc[r] - mx); sum += ex[r]; }
      sum += __shfl_xor(sum, 16, 32);
      const float inv = 1.0f / sum;
      float o = 0.0f;
#pragma unroll
      for (int r = 0; r < 8; ++r)
        o += vpr[wave][(r + sel * 8) * CC + c] * (ex[r] * inv);
      o += __shfl_xor(o, 16, 32);
      if (sel == 0) hout[(size_t)qi * CC + c] = o;
    }
  }
}

// ---------------------------------------------------------------------------
// Kernel 4: out = relu(bn3(relu(bn2(h)) @ w3) + x)
// ---------------------------------------------------------------------------
__global__ void __launch_bounds__(256)
tail_kernel(const float* __restrict__ hin, const float* __restrict__ xin,
            const float* __restrict__ bn2_g, const float* __restrict__ bn2_b,
            const float* __restrict__ w3,
            const float* __restrict__ bn3_g, const float* __restrict__ bn3_b,
            float* __restrict__ out) {
  __shared__ alignas(32) _Float16 wf[4096];
  __shared__ float sc2[CC], of2[CC];
  fill_wfrag(wf, w3);
  const float rs = rsqrtf(1.0f + EPSBN);
  for (int c = threadIdx.x; c < CC; c += blockDim.x) {
    sc2[c] = bn2_g[c] * rs; of2[c] = bn2_b[c];
  }
  __syncthreads();

  const int wave = threadIdx.x >> 5, lane = threadIdx.x & 31;
  const int m = lane & 15, sel = lane >> 4;
  const int rt0 = (blockIdx.x * 8 + wave) * 16;

  hv16 a0, a1;
#pragma unroll
  for (int kt = 0; kt < 2; ++kt) {
    const int k0 = kt * 32;
    const float* rp = hin + (size_t)(rt0 + m) * CC + k0 + sel * 8;
    hv16 a;
#pragma unroll
    for (int t = 0; t < 8; ++t) {
      const int kA = k0 + sel * 8 + t, kB = kA + 16;
      a[t]     = (_Float16)fmaxf(fmaf(rp[t],      sc2[kA], of2[kA]), 0.0f);
      a[8 + t] = (_Float16)fmaxf(fmaf(rp[16 + t], sc2[kB], of2[kB]), 0.0f);
    }
    if (kt == 0) a0 = a; else a1 = a;
  }
#pragma unroll
  for (int nt = 0; nt < 4; ++nt) {
    fv8 acc = {};
    acc = WMMA_F16(a0, ld_b_frag(wf, 0, nt), acc);
    acc = WMMA_F16(a1, ld_b_frag(wf, 1, nt), acc);
    const int c = nt * 16 + m;
    const float s3 = bn3_g[c] * rs, o3 = bn3_b[c];
#pragma unroll
    for (int r = 0; r < 8; ++r) {
      const int row = rt0 + r + sel * 8;
      float v = fmaf(acc[r], s3, o3) + xin[(size_t)row * CC + c];
      out[(size_t)row * CC + c] = fmaxf(v, 0.0f);
    }
  }
}

// ---------------------------------------------------------------------------
extern "C" void kernel_launch(void* const* d_in, const int* in_sizes, int n_in,
                              void* d_out, int out_size, void* d_ws, size_t ws_size,
                              hipStream_t stream) {
  const float* x      = (const float*)d_in[0];
  const float* p      = (const float*)d_in[1];
  const float* w1     = (const float*)d_in[2];
  const float* bn1_g  = (const float*)d_in[3];
  const float* bn1_b  = (const float*)d_in[4];
  const float* wq     = (const float*)d_in[5];
  const float* bq     = (const float*)d_in[6];
  const float* wk     = (const float*)d_in[7];
  const float* bk     = (const float*)d_in[8];
  const float* wv     = (const float*)d_in[9];
  const float* bv     = (const float*)d_in[10];
  const float* lp1_w  = (const float*)d_in[11];
  const float* lp1_b  = (const float*)d_in[12];
  const float* bnp_g  = (const float*)d_in[13];
  const float* bnp_b  = (const float*)d_in[14];
  const float* lp2_w  = (const float*)d_in[15];
  const float* lp2_b  = (const float*)d_in[16];
  const float* bnw1_g = (const float*)d_in[17];
  const float* bnw1_b = (const float*)d_in[18];
  const float* lw1_w  = (const float*)d_in[19];
  const float* lw1_b  = (const float*)d_in[20];
  const float* bnw2_g = (const float*)d_in[21];
  const float* bnw2_b = (const float*)d_in[22];
  const float* lw2_w  = (const float*)d_in[23];
  const float* lw2_b  = (const float*)d_in[24];
  const float* bn2_g  = (const float*)d_in[25];
  const float* bn2_b  = (const float*)d_in[26];
  const float* w3     = (const float*)d_in[27];
  const float* bn3_g  = (const float*)d_in[28];
  const float* bn3_b  = (const float*)d_in[29];

  // d_out = [ out (B,N,C) f32 | p (B,N,3) f32 | ind (B,N,K) i32 ]
  float* outp = (float*)d_out;
  float* pout = outp + (size_t)BB * NN * CC;
  int*   indp = (int*)(outp + (size_t)BB * NN * CC + (size_t)BB * NN * 3);

  // workspace: q, k, v, h  (each B*N*C f32 = 4 MB)
  float* qb = (float*)d_ws;
  float* kb = qb + (size_t)BB * NN * CC;
  float* vb = kb + (size_t)BB * NN * CC;
  float* hb = vb + (size_t)BB * NN * CC;

  knn_kernel<<<BB * (NN / 256), 256, 0, stream>>>(p, indp, pout);
  qkv_kernel<<<(BB * NN / 16) / 8, 256, 0, stream>>>(
      x, w1, bn1_g, bn1_b, wq, bq, wk, bk, wv, bv, qb, kb, vb);
  attn_kernel<<<(BB * NN) / 4, 128, 0, stream>>>(
      p, indp, qb, kb, vb, lp1_w, lp1_b, bnp_g, bnp_b, lp2_w, lp2_b,
      bnw1_g, bnw1_b, lw1_w, lw1_b, bnw2_g, bnw2_b, lw2_w, lw2_b, hb);
  tail_kernel<<<(BB * NN / 16) / 8, 256, 0, stream>>>(
      hb, x, bn2_g, bn2_b, w3, bn3_g, bn3_b, outp);
}
